// Attention3d_9483287790337
// MI455X (gfx1250) — matrix operand order repaired and checked by reading, not compiled, not run on hardware
//
#include <hip/hip_runtime.h>
#include <math.h>

// ---------------------------------------------------------------------------
// Attention3d on MI455X (gfx1250, wave32, WMMA bf16 16x16x32, fp32 accum).
//   0. f2bf / x_pack : one-time bf16 packing (weights, x transpose)
//   1. qkv_gemm      : qkv[768][4096] = w_qkv @ x          (WMMA bf16)
//   2. k_row_stats   : per-k-row max & 1/sum(exp)          (softmax over n)
//   3. softmax_q_pack: softmax over d, *scale, write qT bf16 [h][i][64]
//   4. k_pack        : apply k softmax, transpose -> kT bf16 [h][j][64]
//   5. v_pack        : v/n -> bf16 [h][d][4096]
//   6. flash_attn    : flash attention, K/V tiles staged to LDS with
//                      GLOBAL_LOAD_ASYNC_TO_LDS (double buffered, ASYNCcnt)
//   7. out_ln        : y = w_out@O + b, channel LayerNorm  (WMMA bf16)
// ---------------------------------------------------------------------------

typedef __attribute__((ext_vector_type(16))) __bf16 v16bf;
typedef __attribute__((ext_vector_type(8)))  __bf16 v8bf;
typedef __attribute__((ext_vector_type(2)))  __bf16 v2bf;
typedef __attribute__((ext_vector_type(8)))  float  v8f;
typedef __attribute__((ext_vector_type(4)))  int    v4i;

typedef __attribute__((address_space(1))) v4i v4i_g;   // global int4
typedef __attribute__((address_space(3))) v4i v4i_l;   // LDS int4

#define N_TOK  4096
#define DHEAD  64
#define DIM    256
#define HID    256

__device__ __forceinline__ v16bf join8(v8bf lo, v8bf hi) {
    return __builtin_shufflevector(lo, hi, 0,1,2,3,4,5,6,7,8,9,10,11,12,13,14,15);
}

__device__ __forceinline__ v8f wmma_bf16(v16bf a, v16bf b, v8f c) {
    // D(16x16 f32) = A(16x32 bf16) * B(32x16 bf16) + C
    return __builtin_amdgcn_wmma_f32_16x16x32_bf16(false, a, false, b,
                                                   (short)0, c, false, false);
}

// A-tile (16x32 bf16): lane m = l%16 = row, g = l/16; elements 0..15 =
// K[16g..16g+16)   (contiguous order, matching load_b_bf16).
__device__ __forceinline__ v16bf load_a_bf16(const __bf16* row, int g) {
    return join8(*(const v8bf*)(row + 16*g), *(const v8bf*)(row + 16*g + 8));
}
// B-tile (32x16 bf16): lane n = l%16 = col, elements 0..15 = K[16g..16g+16)
// -> one contiguous 32B run per lane.
__device__ __forceinline__ v16bf load_b_bf16(const __bf16* p) {
    return join8(*(const v8bf*)(p), *(const v8bf*)(p + 8));
}

// ---- CDNA5 async global->LDS copy (16B per lane), ASYNCcnt tracked --------
__device__ __forceinline__ void async_copy_b128(const __bf16* gsrc, __bf16* ldst) {
#if __has_builtin(__builtin_amdgcn_global_load_async_to_lds_b128)
    __builtin_amdgcn_global_load_async_to_lds_b128((v4i_g*)gsrc, (v4i_l*)ldst,
                                                   0, 0);
#else
    unsigned loff = (unsigned)(unsigned long long)ldst;   // flat LDS addr[31:0]
    asm volatile("global_load_async_to_lds_b128 %0, %1, off"
                 :: "v"(loff), "v"(gsrc) : "memory");
#endif
}

template <int N>
__device__ __forceinline__ void wait_asynccnt() {
#if __has_builtin(__builtin_amdgcn_s_wait_asynccnt)
    __builtin_amdgcn_s_wait_asynccnt(N);
#else
    asm volatile("s_wait_asynccnt %0" :: "i"(N) : "memory");
#endif
}

// ---------------------------------------------------------------------------
// 0a) fp32 -> bf16 copy (weight packing)
// ---------------------------------------------------------------------------
__global__ __launch_bounds__(256)
void f2bf(const float* __restrict__ s, __bf16* __restrict__ d, int n) {
    int i = blockIdx.x * 256 + threadIdx.x;
    if (i < n) d[i] = (__bf16)s[i];
}

// ---------------------------------------------------------------------------
// 0b) x[256][4096] fp32 -> xT[4096][256] bf16 (64x64 LDS-tiled transpose)
// ---------------------------------------------------------------------------
__global__ __launch_bounds__(256)
void x_pack(const float* __restrict__ x, __bf16* __restrict__ xT) {
    const int c0 = blockIdx.x * 64;               // 0..3
    const int n0 = blockIdx.y * 64;               // 0..63
    const int col = threadIdx.x & 63, rg = threadIdx.x >> 6;
    __shared__ float tile[64][65];
#pragma unroll 4
    for (int rr = 0; rr < 16; ++rr) {
        int c = rg + 4 * rr;
        tile[c][col] = x[(size_t)(c0 + c) * N_TOK + n0 + col];
    }
    __syncthreads();
#pragma unroll 4
    for (int rr = 0; rr < 16; ++rr) {
        int nn = rg + 4 * rr;
        xT[(size_t)(n0 + nn) * DIM + c0 + col] = (__bf16)tile[col][nn];
    }
}

// ---------------------------------------------------------------------------
// 1) qkv = w_qkv[768][256] @ x[256][4096] (both pre-packed bf16), fp32 out.
//    One wave per 16x16 tile; all fragment loads are contiguous b128.
// ---------------------------------------------------------------------------
__global__ __launch_bounds__(32)
void qkv_gemm(const __bf16* __restrict__ xT, const __bf16* __restrict__ wqb,
              float* __restrict__ qkv) {
    const int l  = threadIdx.x;
    const int lm = l & 15, g = l >> 4;
    const int m0 = blockIdx.x * 16;               // 0..47
    const int n0 = blockIdx.y * 16;               // 0..255

    v8f acc = {};
#pragma unroll
    for (int kt = 0; kt < 8; ++kt) {
        v16bf A = load_a_bf16(wqb + (size_t)(m0 + lm) * DIM + kt * 32, g);
        v16bf B = load_b_bf16(xT + (size_t)(n0 + lm) * DIM + kt * 32 + 16 * g);
        acc = wmma_bf16(A, B, acc);
    }
#pragma unroll
    for (int r = 0; r < 8; ++r)
        qkv[(size_t)(m0 + r + 8*g) * N_TOK + n0 + lm] = acc[r];
}

// ---------------------------------------------------------------------------
// 2) k row stats: rows 256..511 of qkv; kstats[r] = {max, 1/sum(exp)}
// ---------------------------------------------------------------------------
__global__ __launch_bounds__(256)
void k_row_stats(const float* __restrict__ qkv, float* __restrict__ kstats) {
    const int r = blockIdx.x;
    const float* row = qkv + (size_t)(DIM + r) * N_TOK;
    __shared__ float red[256];
    const int t = threadIdx.x;

    float mx = -INFINITY;
    for (int s = t; s < N_TOK; s += 256) mx = fmaxf(mx, row[s]);
    red[t] = mx; __syncthreads();
    for (int off = 128; off > 0; off >>= 1) {
        if (t < off) red[t] = fmaxf(red[t], red[t + off]);
        __syncthreads();
    }
    mx = red[0]; __syncthreads();

    float sm = 0.f;
    for (int s = t; s < N_TOK; s += 256) sm += __expf(row[s] - mx);
    red[t] = sm; __syncthreads();
    for (int off = 128; off > 0; off >>= 1) {
        if (t < off) red[t] += red[t + off];
        __syncthreads();
    }
    if (t == 0) { kstats[2*r] = mx; kstats[2*r + 1] = 1.0f / red[0]; }
}

// ---------------------------------------------------------------------------
// 3) q softmax over d (64) per (h,i), fold *scale, qT bf16 [h*4096+i][64]
// ---------------------------------------------------------------------------
__global__ __launch_bounds__(256)
void softmax_q_pack(const float* __restrict__ qkv, __bf16* __restrict__ qT) {
    const int wid = blockIdx.x * 8 + (threadIdx.x >> 5);   // h*4096 + i
    const int l   = threadIdx.x & 31;
    const int h   = wid >> 12, i = wid & 4095;

    const float* base = qkv + (size_t)(h * DHEAD) * N_TOK + i;
    float a = base[(2*l)     * N_TOK];
    float b = base[(2*l + 1) * N_TOK];

    float mx = fmaxf(a, b);
#pragma unroll
    for (int m = 1; m < 32; m <<= 1) mx = fmaxf(mx, __shfl_xor(mx, m, 32));
    float ea = __expf(a - mx), eb = __expf(b - mx);
    float s = ea + eb;
#pragma unroll
    for (int m = 1; m < 32; m <<= 1) s += __shfl_xor(s, m, 32);
    const float inv = 0.125f / s;                 // SCALE = 64^-0.5 folded in

    v2bf p; p[0] = (__bf16)(ea * inv); p[1] = (__bf16)(eb * inv);
    *(v2bf*)(qT + (size_t)wid * DHEAD + 2*l) = p;
}

// ---------------------------------------------------------------------------
// 4) k: apply softmax-over-n, transpose 64x64 tiles -> kT bf16 [h][j][64]
// ---------------------------------------------------------------------------
__global__ __launch_bounds__(256)
void k_pack(const float* __restrict__ qkv, const float* __restrict__ kstats,
            __bf16* __restrict__ kT) {
    const int h  = blockIdx.x;
    const int i0 = blockIdx.y * 64;
    const int col = threadIdx.x & 63, rg = threadIdx.x >> 6;
    __shared__ float tile[64][65];

#pragma unroll 4
    for (int rr = 0; rr < 16; ++rr) {
        int d   = rg + 4 * rr;
        int row = h * DHEAD + d;
        float v = qkv[(size_t)(DIM + row) * N_TOK + i0 + col];
        tile[d][col] = __expf(v - kstats[2*row]) * kstats[2*row + 1];
    }
    __syncthreads();
#pragma unroll 4
    for (int rr = 0; rr < 16; ++rr) {
        int ii = rg + 4 * rr;
        kT[((size_t)(h * N_TOK) + i0 + ii) * DHEAD + col] = (__bf16)tile[col][ii];
    }
}

// ---------------------------------------------------------------------------
// 5) v: scale by 1/n, convert to bf16, same [h][d][4096] layout
// ---------------------------------------------------------------------------
__global__ __launch_bounds__(256)
void v_pack(const float* __restrict__ qkv, __bf16* __restrict__ vb) {
    const int idx = blockIdx.x * 256 + threadIdx.x;       // < 1048576
    vb[idx] = (__bf16)(qkv[2 * DIM * N_TOK + idx] * (1.0f / 4096.0f));
}

// ---------------------------------------------------------------------------
// 6) flash attention. Block = 4 waves over a 64-row i-block; per 32-wide
//    j-tile the block async-copies K (32x64) and V (64x32) bf16 tiles into
//    LDS (double buffered, ASYNCcnt), then each wave: S = Qt*K (4 WMMA),
//    online softmax (16-lane shfl), P staged C->A layout via LDS, O += P*V.
// ---------------------------------------------------------------------------
__device__ __forceinline__ void issue_tile(const __bf16* kbase, const __bf16* vbase,
                                           int j0, __bf16 (*kl)[64],
                                           __bf16 (*vl)[32], int t) {
    const __bf16* ksrc = kbase + (size_t)j0 * DHEAD;      // 32 rows x 128B, contig
    async_copy_b128(ksrc + t * 8,         &kl[0][0] + t * 8);
    async_copy_b128(ksrc + (t + 128) * 8, &kl[0][0] + (t + 128) * 8);
    const int d0 = t >> 2, p = (t & 3) * 8;               // 64 rows x 64B
    async_copy_b128(vbase + (size_t)d0        * N_TOK + j0 + p, &vl[d0][p]);
    async_copy_b128(vbase + (size_t)(d0 + 32) * N_TOK + j0 + p, &vl[d0 + 32][p]);
}

__global__ __launch_bounds__(128)
void flash_attn(const __bf16* __restrict__ qT, const __bf16* __restrict__ kT,
                const __bf16* __restrict__ vb, __bf16* __restrict__ Ob) {
    const int mb = blockIdx.x;                    // 0..63
    const int h  = blockIdx.y;                    // 0..3
    const int w  = threadIdx.x >> 5;
    const int l  = threadIdx.x & 31;
    const int lm = l & 15, g = l >> 4;
    const int i0 = mb * 64 + w * 16;

    __shared__ __bf16 klds[2][32][DHEAD] __attribute__((aligned(16)));  // 8 KB
    __shared__ __bf16 vlds[2][DHEAD][32] __attribute__((aligned(16)));  // 8 KB
    __shared__ __bf16 plds[4][16][32]    __attribute__((aligned(16)));  // 4 KB

    const __bf16* qrow = qT + ((size_t)(h * N_TOK) + i0 + lm) * DHEAD;
    const v16bf aQ0 = load_a_bf16(qrow,      g);          // K = d 0..31
    const v16bf aQ1 = load_a_bf16(qrow + 32, g);          // K = d 32..63

    const __bf16* kbase = kT + (size_t)h * N_TOK * DHEAD;
    const __bf16* vbase = vb + (size_t)h * DHEAD * N_TOK;

    float mrow[8], lrow[8];
    v8f o[4] = {{}, {}, {}, {}};
#pragma unroll
    for (int r = 0; r < 8; ++r) { mrow[r] = -INFINITY; lrow[r] = 0.f; }

    issue_tile(kbase, vbase, 0, klds[0], vlds[0], threadIdx.x);

    for (int jt = 0; jt < 128; ++jt) {
        const int b = jt & 1;
        if (jt + 1 < 128) {
            issue_tile(kbase, vbase, (jt + 1) * 32, klds[b ^ 1], vlds[b ^ 1],
                       threadIdx.x);
            wait_asynccnt<4>();      // our 4 ops for tile jt have landed
        } else {
            wait_asynccnt<0>();
        }
        __syncthreads();             // all waves' quarters of tile jt visible

        // ---- S = Qt * K from LDS
        const __bf16* kl0 = &klds[b][lm][0];
        const __bf16* kl1 = &klds[b][16 + lm][0];
        v8f S0 = {}, S1 = {};
        S0 = wmma_bf16(aQ0, load_b_bf16(kl0 + 16*g),      S0);
        S0 = wmma_bf16(aQ1, load_b_bf16(kl0 + 32 + 16*g), S0);
        S1 = wmma_bf16(aQ0, load_b_bf16(kl1 + 16*g),      S1);
        S1 = wmma_bf16(aQ1, load_b_bf16(kl1 + 32 + 16*g), S1);

        // ---- online softmax (row r lives at m = r + 8g, cols across 16 lanes)
        float P0[8], P1[8];
#pragma unroll
        for (int r = 0; r < 8; ++r) {
            float t = fmaxf(S0[r], S1[r]);
            t = fmaxf(t, __shfl_xor(t, 1, 32));
            t = fmaxf(t, __shfl_xor(t, 2, 32));
            t = fmaxf(t, __shfl_xor(t, 4, 32));
            t = fmaxf(t, __shfl_xor(t, 8, 32));
            float mn    = fmaxf(mrow[r], t);
            float alpha = __expf(mrow[r] - mn);
            mrow[r] = mn;
            P0[r] = __expf(S0[r] - mn);
            P1[r] = __expf(S1[r] - mn);
            float rs = P0[r] + P1[r];
            rs += __shfl_xor(rs, 1, 32);
            rs += __shfl_xor(rs, 2, 32);
            rs += __shfl_xor(rs, 4, 32);
            rs += __shfl_xor(rs, 8, 32);
            lrow[r] = lrow[r] * alpha + rs;
            o[0][r] *= alpha; o[1][r] *= alpha; o[2][r] *= alpha; o[3][r] *= alpha;
        }

        // ---- stage P through wave-private LDS (C-layout -> A-layout);
        //      per-wave LDS ops execute in order, so no block barrier needed.
#pragma unroll
        for (int r = 0; r < 8; ++r) {
            int m = r + 8*g;
            plds[w][m][lm]      = (__bf16)P0[r];
            plds[w][m][16 + lm] = (__bf16)P1[r];
        }
        const v16bf aP = load_a_bf16(&plds[w][lm][0], g);

        // ---- O += P * V from LDS
#pragma unroll
        for (int dc = 0; dc < 4; ++dc)
            o[dc] = wmma_bf16(aP, load_b_bf16(&vlds[b][dc*16 + lm][16*g]), o[dc]);

        __syncthreads();             // buffer b fully consumed before reuse
    }

    // ---- normalize & write O bf16, token-major [i][h*64+d]
#pragma unroll
    for (int r = 0; r < 8; ++r) {
        const float inv = 1.0f / lrow[r];
        const int m = r + 8*g;
        __bf16* orow = Ob + (size_t)(i0 + m) * HID + h * DHEAD;
#pragma unroll
        for (int dc = 0; dc < 4; ++dc)
            orow[dc*16 + lm] = (__bf16)(o[dc][r] * inv);
    }
}

// ---------------------------------------------------------------------------
// 7) y = w_out @ O + b_out, then channel LayerNorm * g. One wave / 16 tokens.
// ---------------------------------------------------------------------------
__global__ __launch_bounds__(32)
void out_ln(const __bf16* __restrict__ Ob, const __bf16* __restrict__ wob,
            const float* __restrict__ b_out, const float* __restrict__ gg,
            float* __restrict__ out) {
    const int l  = threadIdx.x;
    const int lm = l & 15, g = l >> 4;
    const int i0 = blockIdx.x * 16;
    __shared__ float ylds[256][16];

    v16bf bO[8];
    const __bf16* orow = Ob + (size_t)(i0 + lm) * HID;
#pragma unroll
    for (int kt = 0; kt < 8; ++kt)
        bO[kt] = load_b_bf16(orow + kt*32 + 16*g);

    for (int oc = 0; oc < 16; ++oc) {
        v8f acc = {};
#pragma unroll
        for (int kt = 0; kt < 8; ++kt)
            acc = wmma_bf16(load_a_bf16(wob + (size_t)(oc*16 + lm) * HID + kt*32, g),
                            bO[kt], acc);
#pragma unroll
        for (int r = 0; r < 8; ++r) {
            int o = oc*16 + r + 8*g;
            ylds[o][lm] = acc[r] + b_out[o];
        }
    }
    __syncthreads();

    float sum = 0.f, sq = 0.f;
    for (int o = 0; o < 256; ++o) {
        float v = ylds[o][lm];
        sum += v; sq += v * v;
    }
    const float mean = sum * (1.0f / 256.0f);
    const float var  = sq * (1.0f / 256.0f) - mean * mean;
    const float rstd = rsqrtf(var + 1e-5f);

    for (int ob = 0; ob < 128; ++ob) {
        int o = 2*ob + g;
        out[(size_t)o * N_TOK + i0 + lm] = (ylds[o][lm] - mean) * rstd * gg[o];
    }
}

// ---------------------------------------------------------------------------
// Host launcher
// ---------------------------------------------------------------------------
extern "C" void kernel_launch(void* const* d_in, const int* in_sizes, int n_in,
                              void* d_out, int out_size, void* d_ws, size_t ws_size,
                              hipStream_t stream) {
    const float* x     = (const float*)d_in[0];   // [256][4096]
    const float* w_qkv = (const float*)d_in[1];   // [768][256]
    const float* w_out = (const float*)d_in[2];   // [256][256]
    const float* b_out = (const float*)d_in[3];   // [256]
    const float* g     = (const float*)d_in[4];   // [256]
    float* out = (float*)d_out;                   // [256][4096]

    char* ws = (char*)d_ws;
    size_t off = 0;
    float*  qkv    = (float*)(ws + off); off += (size_t)768 * 4096 * 4;   // 12 MB
    float*  kstats = (float*)(ws + off); off += 256 * 2 * 4;
    __bf16* qT     = (__bf16*)(ws + off); off += (size_t)4 * 4096 * 64 * 2;
    __bf16* kT     = (__bf16*)(ws + off); off += (size_t)4 * 4096 * 64 * 2;
    __bf16* vb     = (__bf16*)(ws + off); off += (size_t)4 * 4096 * 64 * 2;
    __bf16* Ob     = (__bf16*)(ws + off); off += (size_t)4096 * 256 * 2;
    __bf16* xT     = (__bf16*)(ws + off); off += (size_t)4096 * 256 * 2;
    __bf16* wqb    = (__bf16*)(ws + off); off += (size_t)768 * 256 * 2;
    __bf16* wob    = (__bf16*)(ws + off); off += (size_t)256 * 256 * 2;

    f2bf          <<<768,           256, 0, stream>>>(w_qkv, wqb, 768 * 256);
    f2bf          <<<256,           256, 0, stream>>>(w_out, wob, 256 * 256);
    x_pack        <<<dim3(4, 64),   256, 0, stream>>>(x, xT);
    qkv_gemm      <<<dim3(48, 256), 32,  0, stream>>>(xT, wqb, qkv);
    k_row_stats   <<<256,           256, 0, stream>>>(qkv, kstats);
    softmax_q_pack<<<2048,          256, 0, stream>>>(qkv, qT);
    k_pack        <<<dim3(4, 64),   256, 0, stream>>>(qkv, kstats, kT);
    v_pack        <<<4096,          256, 0, stream>>>(qkv, vb);
    flash_attn    <<<dim3(64, 4),   128, 0, stream>>>(qT, kT, vb, Ob);
    out_ln        <<<256,           32,  0, stream>>>(Ob, wob, b_out, g, out);
}
